// MLPNNodeDecoder_23733989278149
// MI455X (gfx1250) — matrix operand order repaired and checked by reading, not compiled, not run on hardware
//
#include <hip/hip_runtime.h>
#include <hip/hip_bf16.h>

typedef __attribute__((ext_vector_type(16))) _Float16 v16h;
typedef __attribute__((ext_vector_type(8)))  _Float16 v8h;
typedef __attribute__((ext_vector_type(8)))  float    v8f;

#define NUM_NODES 116
#define EMBED 64
#define KSPLIT 4

__device__ __forceinline__ float mishf(float x) {
    float sp = log1pf(expf(x));
    return x * tanhf(sp);
}

// ---------------- zero ----------------
__global__ void zero_f32(float* __restrict__ p, int n) {
    int t = blockIdx.x * blockDim.x + threadIdx.x;
    if (t < n) p[t] = 0.0f;
}

// ---------------- edge degree count ----------------
__global__ void edge_cnt(const long long* __restrict__ dst, float* __restrict__ cnt, int E) {
    int t = blockIdx.x * blockDim.x + threadIdx.x;
    if (t < E) atomicAdd(&cnt[dst[t]], 1.0f);
}

// ---------------- scatter-add of source features ----------------
__global__ void edge_agg(const float* __restrict__ x,
                         const long long* __restrict__ src,
                         const long long* __restrict__ dst,
                         float* __restrict__ agg, int E) {
    int e = blockIdx.x * blockDim.x + threadIdx.x;
    if (e >= E) return;
    long long s = src[e], d = dst[e];
    const float* xr = x + s * EMBED;
    float* ar = agg + d * EMBED;
#pragma unroll
    for (int c = 0; c < EMBED; c += 4) {
        float4 v = *(const float4*)(xr + c);
        atomicAdd(ar + c + 0, v.x);
        atomicAdd(ar + c + 1, v.y);
        atomicAdd(ar + c + 2, v.z);
        atomicAdd(ar + c + 3, v.w);
    }
}

// ------- SAGE linear: out = mish(mean @ wl.T + bl + x @ wr.T) -------
__global__ void sage_lin(const float* __restrict__ x, const float* __restrict__ agg,
                         const float* __restrict__ cnt,
                         const float* __restrict__ wl, const float* __restrict__ bl,
                         const float* __restrict__ wr,
                         float* __restrict__ out, int Nn) {
    int t = blockIdx.x * blockDim.x + threadIdx.x;
    if (t >= Nn * EMBED) return;
    int n = t >> 6, o = t & 63;
    float c = cnt[n];
    float inv = 1.0f / (c < 1.0f ? 1.0f : c);
    const float* xr  = x   + (size_t)n * EMBED;
    const float* ar  = agg + (size_t)n * EMBED;
    const float* wlr = wl  + (size_t)o * EMBED;
    const float* wrr = wr  + (size_t)o * EMBED;
    float acc = bl[o];
#pragma unroll 8
    for (int k = 0; k < EMBED; ++k)
        acc = fmaf(ar[k] * inv, wlr[k], fmaf(xr[k], wrr[k], acc));
    out[t] = mishf(acc);
}

// ---------------- f32 -> f16 ----------------
__global__ void cvt_f16(const float* __restrict__ in, _Float16* __restrict__ out, int n) {
    int t = blockIdx.x * blockDim.x + threadIdx.x;
    if (t < n) out[t] = (_Float16)in[t];
}

// ------- WMMA NT GEMM (split-K partials) -------
// part[z][m][n] = A[m, kz..kz+K/KS) @ Bw[n, kz..)^T   (no bias/act here)
// One wave computes a 16x16 tile of one K-chunk; 8 waves/block = 8 N-tiles.
__global__ __launch_bounds__(256)
void wmma_gemm_nt_part(const _Float16* __restrict__ A, const float* __restrict__ Bw,
                       float* __restrict__ part, int K, int Nout, int Mtot) {
    const int lane = threadIdx.x & 31;
    const int wave = threadIdx.x >> 5;
    const int l16  = lane & 15;
    const int hi   = lane >> 4;          // half-wave select
    const int nt   = blockIdx.x * 8 + wave;
    if (nt * 16 >= Nout) return;
    const int n0 = nt * 16;
    const int m0 = blockIdx.y * 16;
    const int kc   = K / KSPLIT;
    const int kbeg = blockIdx.z * kc;
    const int kend = kbeg + kc;

    const _Float16* Ar = A  + (size_t)(m0 + l16) * K;   // A: lane&15 = row M
    const float*    Br = Bw + (size_t)(n0 + l16) * K;   // B: lane&15 = col N

    v8f acc = {};
#pragma unroll 2
    for (int k0 = kbeg; k0 < kend; k0 += 32) {
        // A fragment (contiguous k order, same as B): lanes 0-15 K=0..15, lanes 16-31 K=16..31
        v8h alo = *(const v8h*)(Ar + k0 + hi * 16);
        v8h ahi = *(const v8h*)(Ar + k0 + hi * 16 + 8);
        v16h a  = __builtin_shufflevector(alo, ahi, 0,1,2,3,4,5,6,7,8,9,10,11,12,13,14,15);

        // B fragment (contiguous k order, same as A): lane&15 = col, half-wave holds K 0..15 / 16..31
        const float4* Bp = (const float4*)(Br + k0 + hi * 16);
        __builtin_prefetch(Br + k0 + 2048, 0, 3);   // global_prefetch_b8, ~8KB ahead, toward WGP scope
        float4 w0 = Bp[0], w1 = Bp[1], w2 = Bp[2], w3 = Bp[3];
        v16h b = { (_Float16)w0.x, (_Float16)w0.y, (_Float16)w0.z, (_Float16)w0.w,
                   (_Float16)w1.x, (_Float16)w1.y, (_Float16)w1.z, (_Float16)w1.w,
                   (_Float16)w2.x, (_Float16)w2.y, (_Float16)w2.z, (_Float16)w2.w,
                   (_Float16)w3.x, (_Float16)w3.y, (_Float16)w3.z, (_Float16)w3.w };

        acc = __builtin_amdgcn_wmma_f32_16x16x32_f16(
                  false, a, false, b, (short)0, acc, false, false);
    }

    float* pp = part + (size_t)blockIdx.z * Mtot * Nout;
#pragma unroll
    for (int r = 0; r < 8; ++r) {
        int m = m0 + hi * 8 + r;                // C/D: VGPR r -> row r (+8 for upper half)
        pp[(size_t)m * Nout + n0 + l16] = acc[r];
    }
}

// ------- reduce split-K partials + bias (+mish) -------
template <bool ACT, typename OutT>
__global__ void reduce_part(const float* __restrict__ part, const float* __restrict__ bias,
                            OutT* __restrict__ out, int Nout, int total) {
    int t = blockIdx.x * blockDim.x + threadIdx.x;
    if (t >= total) return;
    int n = t % Nout;
    float s = bias[n];
#pragma unroll
    for (int z = 0; z < KSPLIT; ++z) s += part[(size_t)z * total + t];
    if (ACT) s = mishf(s);
    out[t] = (OutT)s;
}

// ------- symmetrize + unit diagonal -------
__global__ void sym_out(const float* __restrict__ xrec, float* __restrict__ out, int total) {
    int t = blockIdx.x * blockDim.x + threadIdx.x;
    if (t >= total) return;
    int j   = t % NUM_NODES;
    int row = t / NUM_NODES;
    int i   = row % NUM_NODES;
    int b   = row / NUM_NODES;
    const float* xb = xrec + (size_t)b * NUM_NODES * NUM_NODES;
    out[t] = (i == j) ? 1.0f
                      : 0.5f * (xb[i * NUM_NODES + j] + xb[j * NUM_NODES + i]);
}

static inline char* align256(char* p) {
    return (char*)(((uintptr_t)p + 255) & ~(uintptr_t)255);
}

extern "C" void kernel_launch(void* const* d_in, const int* in_sizes, int n_in,
                              void* d_out, int out_size, void* d_ws, size_t ws_size,
                              hipStream_t stream) {
    const float*      z     = (const float*)d_in[0];
    const long long*  ei    = (const long long*)d_in[1];
    const float*      w1_l  = (const float*)d_in[3];
    const float*      b1_l  = (const float*)d_in[4];
    const float*      w1_r  = (const float*)d_in[5];
    const float*      w2_l  = (const float*)d_in[6];
    const float*      b2_l  = (const float*)d_in[7];
    const float*      w2_r  = (const float*)d_in[8];
    const float*      fc1_w = (const float*)d_in[9];
    const float*      fc1_b = (const float*)d_in[10];
    const float*      fc2_w = (const float*)d_in[11];
    const float*      fc2_b = (const float*)d_in[12];

    const int Nn      = in_sizes[0] / EMBED;     // 7424 nodes
    const int E       = in_sizes[1] / 2;         // 237568 edges
    const int B       = Nn / NUM_NODES;          // 64 graphs
    const int d_hid   = in_sizes[10];            // 7424
    const int d_out_f = in_sizes[12];            // 13456
    const int Kfc     = in_sizes[9] / d_hid;     // 7424 (fc1 fan-in)
    const long long* src = ei;
    const long long* dst = ei + E;

    // workspace carve
    char* w = (char*)d_ws;
    float* agg  = (float*)w;            w = align256(w + (size_t)Nn * EMBED * 4);
    float* cnt  = (float*)w;            w = align256(w + (size_t)Nn * 4);
    float* h1   = (float*)w;            w = align256(w + (size_t)Nn * EMBED * 4);
    float* h2   = (float*)w;            w = align256(w + (size_t)Nn * EMBED * 4);
    _Float16* h2h  = (_Float16*)w;      w = align256(w + (size_t)Nn * EMBED * 2);
    _Float16* hidh = (_Float16*)w;      w = align256(w + (size_t)B * d_hid * 2);
    float* xrec = (float*)w;            w = align256(w + (size_t)B * d_out_f * 4);
    const int maxN = d_hid > d_out_f ? d_hid : d_out_f;
    float* partbuf = (float*)w;         w = align256(w + (size_t)KSPLIT * B * maxN * 4);

    const int TB = 256;
    const int nf = Nn * EMBED;

    // ---- degree counts (shared by both SAGE layers) ----
    zero_f32<<<(Nn + TB - 1) / TB, TB, 0, stream>>>(cnt, Nn);
    edge_cnt<<<(E + TB - 1) / TB, TB, 0, stream>>>(dst, cnt, E);

    // ---- SAGE layer 1 ----
    zero_f32<<<(nf + TB - 1) / TB, TB, 0, stream>>>(agg, nf);
    edge_agg<<<(E + TB - 1) / TB, TB, 0, stream>>>(z, src, dst, agg, E);
    sage_lin<<<(nf + TB - 1) / TB, TB, 0, stream>>>(z, agg, cnt, w1_l, b1_l, w1_r, h1, Nn);

    // ---- SAGE layer 2 ----
    zero_f32<<<(nf + TB - 1) / TB, TB, 0, stream>>>(agg, nf);
    edge_agg<<<(E + TB - 1) / TB, TB, 0, stream>>>(h1, src, dst, agg, E);
    sage_lin<<<(nf + TB - 1) / TB, TB, 0, stream>>>(h1, agg, cnt, w2_l, b2_l, w2_r, h2, Nn);

    // ---- convert activations to f16 (h2 viewed as (B, NUM_NODES*EMBED) row-major) ----
    cvt_f16<<<(nf + TB - 1) / TB, TB, 0, stream>>>(h2, h2h, nf);

    // ---- fc1: hidden(f16) = mish(h2h @ fc1_w^T + fc1_b), M=B, N=d_hid, K=Kfc ----
    {
        dim3 grid(((d_hid + 15) / 16 + 7) / 8, B / 16, KSPLIT);
        wmma_gemm_nt_part<<<grid, 256, 0, stream>>>(h2h, fc1_w, partbuf, Kfc, d_hid, B);
        reduce_part<true, _Float16><<<(B * d_hid + TB - 1) / TB, TB, 0, stream>>>(
            partbuf, fc1_b, hidh, d_hid, B * d_hid);
    }

    // ---- fc2: xrec(f32) = hidh @ fc2_w^T + fc2_b, M=B, N=d_out_f, K=d_hid ----
    {
        dim3 grid(((d_out_f + 15) / 16 + 7) / 8, B / 16, KSPLIT);
        wmma_gemm_nt_part<<<grid, 256, 0, stream>>>(hidh, fc2_w, partbuf, d_hid, d_out_f, B);
        reduce_part<false, float><<<(B * d_out_f + TB - 1) / TB, TB, 0, stream>>>(
            partbuf, fc2_b, xrec, d_out_f, B * d_out_f);
    }

    // ---- symmetrize + unit diagonal ----
    sym_out<<<(out_size + TB - 1) / TB, TB, 0, stream>>>(xrec, (float*)d_out, out_size);
}
